// GNN_62569083568974
// MI455X (gfx1250) — matrix operand order repaired and checked by reading, not compiled, not run on hardware
//
#include <hip/hip_runtime.h>
#include <hip/hip_bf16.h>
#include <stdint.h>

#define N_NODES  20000
#define N_EDGES  200000
#define NODE_DIM 32
#define EDGE_DIM 16
#define HIDDEN   32
#define N_GRAPHS 64

#define NTILES   (N_EDGES / 16)      // 12500 edge tiles of 16
#define KSTEPS   33                  // 32 weight K-steps + 1 bias step
#define LDS_KSTEPS 32                // weight K-steps staged in LDS (64 KB)
#define BPK_ELEMS (KSTEPS * 2 * 32 * 16)      // packed B fragments (f16)
#define LDS_ELEMS (LDS_KSTEPS * 2 * 32 * 16)  // 32768 f16 = 65536 B

typedef __attribute__((ext_vector_type(16))) _Float16 v16h;
typedef __attribute__((ext_vector_type(8)))  float    v8f;

// ---------------- utility kernels ----------------

__global__ void zero_u32(uint32_t* p, int n) {
    int i = blockIdx.x * blockDim.x + threadIdx.x;
    if (i < n) p[i] = 0u;
}

__global__ void count_kernel(const int* __restrict__ ids, int n, int* __restrict__ cnt) {
    int i = blockIdx.x * blockDim.x + threadIdx.x;
    if (i < n) atomicAdd(&cnt[ids[i]], 1);
}

// Pack W2r[(k,i),o] (+ bias b2 as extra K-step 32) into the per-lane WMMA B
// fragment layout: element j of lane L in fragment (t, ntile) holds
// B[K_local = (L>>4)*16 + j, N = (L&15)] for global kappa = 32*t + K_local,
// where W2r[kappa, o] = W2[kappa>>5, (kappa&31)*32 + o].
__global__ void pack_w2(const float* __restrict__ W2, const float* __restrict__ b2,
                        _Float16* __restrict__ Bpk) {
    int idx = blockIdx.x * blockDim.x + threadIdx.x;
    if (idx >= BPK_ELEMS) return;
    int j    = idx & 15;
    int lane = (idx >> 4) & 31;
    int nt   = (idx >> 9) & 1;
    int t    = idx >> 10;
    int half = lane >> 4;
    int n    = lane & 15;
    int o    = nt * 16 + n;
    int kl   = half * 16 + j;                 // K_local within this 32-slice
    float v = (t < 32) ? W2[t * (NODE_DIM * HIDDEN) + kl * HIDDEN + o]
                       : b2[kl * HIDDEN + o];
    Bpk[idx] = (_Float16)v;
}

// Edge MLP stage 1: H[e,k] = relu(edge_attr[e,:] @ W1 + b1)  (K=16, trivial on VALU)
__global__ void edge_hidden(const float* __restrict__ EA, const float* __restrict__ W1,
                            const float* __restrict__ b1, float* __restrict__ H) {
    int tid = blockIdx.x * blockDim.x + threadIdx.x;
    if (tid >= N_EDGES * HIDDEN) return;
    int e = tid >> 5, k = tid & 31;
    float s = b1[k];
#pragma unroll
    for (int j = 0; j < EDGE_DIM; ++j)
        s += EA[e * EDGE_DIM + j] * W1[j * HIDDEN + k];
    H[tid] = s > 0.f ? s : 0.f;
}

// ---------------- core WMMA edge-message GEMM ----------------
// One wave per 16-edge tile. Computes D[16,32] = Z[16,1024+32] @ [W2r ; b2r]
// with Z built on the fly: row m of Z at K-step t is h[m][t] * x_src[m][0..31]
// (bias step uses A = x itself). The shared weight-fragment table (64 KB) is
// staged into LDS once per workgroup; hot loop reads it with ds_load_b128.
__global__ __launch_bounds__(256) void msg_wmma(
        const float* __restrict__ H, const float* __restrict__ X,
        const int* __restrict__ src, const int* __restrict__ tgt,
        const _Float16* __restrict__ Bpk, float* __restrict__ msg) {
    __shared__ __align__(16) _Float16 Bl[LDS_ELEMS];   // 64 KB of 320 KB/WGP

    // cooperative stage: 65536 B / 256 threads = 16 x uint4 per thread
    {
        const uint4* sv = (const uint4*)Bpk;
        uint4*       dv = (uint4*)Bl;
#pragma unroll
        for (int i = 0; i < (LDS_ELEMS * 2 / 16) / 256; ++i)
            dv[threadIdx.x + i * 256] = sv[threadIdx.x + i * 256];
    }
    __syncthreads();   // all waves reach this (no early return above)

    int wave = threadIdx.x >> 5;
    int lane = threadIdx.x & 31;
    int tile = blockIdx.x * 8 + wave;

    if (tile < NTILES) {               // wave-uniform; EXEC all-1 inside
        int e0   = tile * 16;
        int m    = lane & 15;
        int half = lane >> 4;

        int s = src[e0 + m];

        // x fragment, packed f16 once: contiguous K layout matching B,
        // K = half*16 + j
        v16h xh;
#pragma unroll
        for (int j = 0; j < 16; ++j) {
            int loc = half * 16 + j;
            xh[j] = (_Float16)X[s * HIDDEN + loc];
        }

        // bias-step B fragments (K-step 32) straight from global, held in regs
        const v16h* Bv = (const v16h*)Bpk;
        v16h bb0 = Bv[(32 * 2 + 0) * 32 + lane];
        v16h bb1 = Bv[(32 * 2 + 1) * 32 + lane];

        // per-lane H row for this edge (32 floats, vectorized loads)
        float hrow[32];
        const float4* hp = (const float4*)(H + (size_t)(e0 + m) * HIDDEN);
#pragma unroll
        for (int q = 0; q < 8; ++q) {
            float4 v = hp[q];
            hrow[4 * q + 0] = v.x; hrow[4 * q + 1] = v.y;
            hrow[4 * q + 2] = v.z; hrow[4 * q + 3] = v.w;
        }

        const v16h* Blv = (const v16h*)Bl;
        v8f acc0 = {0.f, 0.f, 0.f, 0.f, 0.f, 0.f, 0.f, 0.f};
        v8f acc1 = acc0;

#pragma unroll
        for (int t = 0; t < LDS_KSTEPS; ++t) {
            _Float16 hsh = (_Float16)hrow[t];
            v16h a = xh * hsh;                     // 8x v_pk_mul_f16
            v16h b0 = Blv[(t * 2 + 0) * 32 + lane]; // ds_load_b128 x2
            v16h b1 = Blv[(t * 2 + 1) * 32 + lane];
            acc0 = __builtin_amdgcn_wmma_f32_16x16x32_f16(false, a, false, b0,
                                                          (short)0, acc0, false, false);
            acc1 = __builtin_amdgcn_wmma_f32_16x16x32_f16(false, a, false, b1,
                                                          (short)0, acc1, false, false);
        }
        // bias K-step: A = x fragment itself
        acc0 = __builtin_amdgcn_wmma_f32_16x16x32_f16(false, xh, false, bb0,
                                                      (short)0, acc0, false, false);
        acc1 = __builtin_amdgcn_wmma_f32_16x16x32_f16(false, xh, false, bb1,
                                                      (short)0, acc1, false, false);

        // D layout: lane holds (M = half*8 + r, N = lane&15) in VGPR r
        int n = lane & 15;
#pragma unroll
        for (int r = 0; r < 8; ++r) {
            int e  = e0 + half * 8 + r;
            int tg = tgt[e];
            atomicAdd(&msg[tg * HIDDEN + n],      acc0[r]);
            atomicAdd(&msg[tg * HIDDEN + 16 + n], acc1[r]);
        }
    }
}

// out[v,o] = relu( msg[v,o]/max(cnt,1) + (X @ root)[v,o] + bias[o] )
__global__ void node_update(const float* __restrict__ msg, const int* __restrict__ cnt,
                            const float* __restrict__ X, const float* __restrict__ root,
                            const float* __restrict__ bias, float* __restrict__ out) {
    int tid = blockIdx.x * blockDim.x + threadIdx.x;
    if (tid >= N_NODES * HIDDEN) return;
    int v = tid >> 5, o = tid & 31;
    int c = cnt[v];
    float s = msg[tid] / (float)(c > 0 ? c : 1) + bias[o];
#pragma unroll
    for (int k = 0; k < HIDDEN; ++k)
        s += X[v * HIDDEN + k] * root[k * HIDDEN + o];
    out[tid] = s > 0.f ? s : 0.f;
}

__global__ void pool_kernel(const float* __restrict__ h, const int* __restrict__ batch,
                            float* __restrict__ pooled) {
    int tid = blockIdx.x * blockDim.x + threadIdx.x;
    if (tid >= N_NODES * HIDDEN) return;
    int v = tid >> 5, o = tid & 31;
    atomicAdd(&pooled[batch[v] * HIDDEN + o], h[tid]);
}

__global__ void head_kernel(const float* __restrict__ pooled, const int* __restrict__ gcnt,
                            const float* __restrict__ fcW, const float* __restrict__ fcb,
                            float* __restrict__ out) {
    int tid = blockIdx.x * blockDim.x + threadIdx.x;
    if (tid >= N_GRAPHS * 2) return;
    int g = tid >> 1, c = tid & 1;
    int gc = gcnt[g];
    float inv = 1.0f / (float)(gc > 0 ? gc : 1);
    float s = fcb[c];
#pragma unroll
    for (int o = 0; o < HIDDEN; ++o)
        s += pooled[g * HIDDEN + o] * inv * fcW[o * 2 + c];
    out[tid] = s;
}

// ---------------- host-side launch sequence ----------------

extern "C" void kernel_launch(void* const* d_in, const int* in_sizes, int n_in,
                              void* d_out, int out_size, void* d_ws, size_t ws_size,
                              hipStream_t stream) {
    (void)in_sizes; (void)n_in; (void)out_size; (void)ws_size;

    const float* x     = (const float*)d_in[0];
    const float* ea    = (const float*)d_in[1];
    const int*   ei    = (const int*)d_in[2];
    const int*   batch = (const int*)d_in[3];
    const float* en1W1 = (const float*)d_in[4];
    const float* en1b1 = (const float*)d_in[5];
    const float* en1W2 = (const float*)d_in[6];
    const float* en1b2 = (const float*)d_in[7];
    const float* root1 = (const float*)d_in[8];
    const float* bias1 = (const float*)d_in[9];
    const float* en2W1 = (const float*)d_in[10];
    const float* en2b1 = (const float*)d_in[11];
    const float* en2W2 = (const float*)d_in[12];
    const float* en2b2 = (const float*)d_in[13];
    const float* root2 = (const float*)d_in[14];
    const float* bias2 = (const float*)d_in[15];
    const float* fcW   = (const float*)d_in[16];
    const float* fcb   = (const float*)d_in[17];
    float* out = (float*)d_out;

    const int* src = ei;             // edge_index[0]
    const int* tgt = ei + N_EDGES;   // edge_index[1]

    // carve scratch (256B-aligned slices)
    char* ws = (char*)d_ws;
    size_t off = 0;
    auto alloc = [&](size_t bytes) -> char* {
        char* p = ws + off;
        off = (off + bytes + 255) & ~(size_t)255;
        return p;
    };
    float*    H      = (float*)alloc((size_t)N_EDGES * HIDDEN * 4);
    float*    msg    = (float*)alloc((size_t)N_NODES * HIDDEN * 4);
    float*    hA     = (float*)alloc((size_t)N_NODES * HIDDEN * 4);
    float*    hB     = (float*)alloc((size_t)N_NODES * HIDDEN * 4);
    int*      cnt    = (int*)alloc((size_t)N_NODES * 4);
    int*      gcnt   = (int*)alloc((size_t)N_GRAPHS * 4);
    float*    pooled = (float*)alloc((size_t)N_GRAPHS * HIDDEN * 4);
    _Float16* Bpk1   = (_Float16*)alloc((size_t)BPK_ELEMS * 2);
    _Float16* Bpk2   = (_Float16*)alloc((size_t)BPK_ELEMS * 2);

    const int TB = 256;
    auto blks = [](int n, int tb) { return (n + tb - 1) / tb; };

    // counts (shared by both layers) + graph counts
    zero_u32<<<blks(N_NODES, TB), TB, 0, stream>>>((uint32_t*)cnt, N_NODES);
    zero_u32<<<1, 64, 0, stream>>>((uint32_t*)gcnt, N_GRAPHS);
    count_kernel<<<blks(N_EDGES, TB), TB, 0, stream>>>(tgt, N_EDGES, cnt);
    count_kernel<<<blks(N_NODES, TB), TB, 0, stream>>>(batch, N_NODES, gcnt);

    // pack weight fragments for both layers
    pack_w2<<<blks(BPK_ELEMS, TB), TB, 0, stream>>>(en1W2, en1b2, Bpk1);
    pack_w2<<<blks(BPK_ELEMS, TB), TB, 0, stream>>>(en2W2, en2b2, Bpk2);

    // ---- layer 1 ----
    edge_hidden<<<blks(N_EDGES * HIDDEN, TB), TB, 0, stream>>>(ea, en1W1, en1b1, H);
    zero_u32<<<blks(N_NODES * HIDDEN, TB), TB, 0, stream>>>((uint32_t*)msg, N_NODES * HIDDEN);
    msg_wmma<<<(NTILES + 7) / 8, 256, 0, stream>>>(H, x, src, tgt, Bpk1, msg);
    node_update<<<blks(N_NODES * HIDDEN, TB), TB, 0, stream>>>(msg, cnt, x, root1, bias1, hA);

    // ---- layer 2 ----
    edge_hidden<<<blks(N_EDGES * HIDDEN, TB), TB, 0, stream>>>(ea, en2W1, en2b1, H);
    zero_u32<<<blks(N_NODES * HIDDEN, TB), TB, 0, stream>>>((uint32_t*)msg, N_NODES * HIDDEN);
    msg_wmma<<<(NTILES + 7) / 8, 256, 0, stream>>>(H, hA, src, tgt, Bpk2, msg);
    node_update<<<blks(N_NODES * HIDDEN, TB), TB, 0, stream>>>(msg, cnt, hA, root2, bias2, hB);

    // ---- pool + head ----
    zero_u32<<<blks(N_GRAPHS * HIDDEN, TB), TB, 0, stream>>>((uint32_t*)pooled, N_GRAPHS * HIDDEN);
    pool_kernel<<<blks(N_NODES * HIDDEN, TB), TB, 0, stream>>>(hB, batch, pooled);
    head_kernel<<<1, 128, 0, stream>>>(pooled, gcnt, fcW, fcb, out);
}
